// ContrastiveLoss_23708219474596
// MI455X (gfx1250) — matrix operand order repaired and checked by reading, not compiled, not run on hardware
//
#include <hip/hip_runtime.h>
#include <hip/hip_bf16.h>

typedef __attribute__((ext_vector_type(16))) _Float16 v16h;
typedef __attribute__((ext_vector_type(8)))  _Float16 v8h;
typedef __attribute__((ext_vector_type(4)))  _Float16 v4h;
typedef __attribute__((ext_vector_type(8)))  float    v8f;

#define B_DIM 4096
#define D_DIM 1024
#define BM 128
#define BN 64
#define BK 32
#define KSTEPS (D_DIM / BK)
#define MARGIN 0.5f
#define EPS 1e-8f

// ---------------------------------------------------------------------------
// CDNA5 async global->LDS copy (16B per lane), tracked by ASYNCcnt.
// LDS destination address comes from the vdst VGPR; low 32 bits of a generic __shared__ pointer
// are the LDS byte offset. GV addressing mode (64-bit VGPR address, SADDR=off).
// ---------------------------------------------------------------------------
__device__ __forceinline__ void async_copy_b128(void* lds_ptr, const void* gptr) {
    unsigned l = (unsigned)(unsigned long long)lds_ptr;
    asm volatile("global_load_async_to_lds_b128 %0, %1, off"
                 :: "v"(l), "v"(gptr)
                 : "memory");
}

__device__ __forceinline__ void wait_asynccnt0() {
    asm volatile("s_wait_asynccnt 0x0" ::: "memory");
}

// ---------------------------------------------------------------------------
// Kernel 1: per-row L2 normalization, f32 -> f16
// grid = 3*4096 blocks (f0 rows, f1 rows, t rows), 256 threads
// ---------------------------------------------------------------------------
__global__ void normalize_rows(const float* __restrict__ f0,
                               const float* __restrict__ f1,
                               const float* __restrict__ t,
                               _Float16* __restrict__ o0,
                               _Float16* __restrict__ o1,
                               _Float16* __restrict__ ot) {
    const int b     = blockIdx.x;
    const int which = b >> 12;          // 0 = f0, 1 = f1, 2 = t
    const int row   = b & (B_DIM - 1);

    const float* src = (which == 0) ? f0 : (which == 1) ? f1 : t;
    _Float16*    dst = (which == 0) ? o0 : (which == 1) ? o1 : ot;

    const float4* s4 = (const float4*)(src + (size_t)row * D_DIM);
    float4 v = s4[threadIdx.x];                 // 256 threads * 4 = 1024
    float ss = v.x * v.x + v.y * v.y + v.z * v.z + v.w * v.w;

    __shared__ float red[256];
    red[threadIdx.x] = ss;
    __syncthreads();
    for (int s = 128; s > 0; s >>= 1) {
        if (threadIdx.x < s) red[threadIdx.x] += red[threadIdx.x + s];
        __syncthreads();
    }
    const float scale = 1.0f / fmaxf(sqrtf(red[0]), EPS);

    v4h hv;
    hv[0] = (_Float16)(v.x * scale);
    hv[1] = (_Float16)(v.y * scale);
    hv[2] = (_Float16)(v.z * scale);
    hv[3] = (_Float16)(v.w * scale);
    ((v4h*)(dst + (size_t)row * D_DIM))[threadIdx.x] = hv;
}

// ---------------------------------------------------------------------------
// Kernel 2: fused dual-GEMM (sim0, sim1 share the B matrix) + margin loss.
// Block tile: 128 (M) x 64 (N). 8 waves of 32; wave w owns M-subtile w and all
// 4 N-subtiles for both sims: 8 x v_wmma_f32_16x16x32_f16 per K-step.
// Double-buffered LDS fed by global_load_async_to_lds_b128 (ASYNCcnt),
// one barrier per K-step.
// ---------------------------------------------------------------------------
__global__ void __launch_bounds__(256)
gemm_loss(const _Float16* __restrict__ f0n,
          const _Float16* __restrict__ f1n,
          const _Float16* __restrict__ tn,
          const float* __restrict__ labels,
          float* __restrict__ partials) {
    __shared__ __align__(64) _Float16 lA0[2][BM * BK];  // 2 x 8 KB
    __shared__ __align__(64) _Float16 lA1[2][BM * BK];  // 2 x 8 KB
    __shared__ __align__(64) _Float16 lB [2][BN * BK];  // 2 x 4 KB
    __shared__ float red[256];

    const int tid  = threadIdx.x;
    const int wave = tid >> 5;          // 0..7 -> M subtile
    const int lane = tid & 31;
    const int rowM0 = blockIdx.y * BM;
    const int colN0 = blockIdx.x * BN;

    const v8f vzero = {0.f, 0.f, 0.f, 0.f, 0.f, 0.f, 0.f, 0.f};
    v8f acc0[4], acc1[4];
#pragma unroll
    for (int nt = 0; nt < 4; ++nt) { acc0[nt] = vzero; acc1[nt] = vzero; }

    // Fragment addressing per ISA 7.12.2 (wave32):
    // A (16-bit, 16x32): lane<16 -> halves {K 0..7, K 16..23}; lane>=16 -> {8..15, 24..31}
    // B (32x16): same K order as A: lane<16 -> {K 0..7, K 16..23}; lane>=16 -> {8..15, 24..31}
    const int mrow  = wave * 16 + (lane & 15);
    const int kA_lo = (lane < 16) ? 0 : 8;
    const int kB    = (lane < 16) ? 0 : 16;

    // Cooperative async-staging indices (A: 2 x 16B/thread, B: 1 x 16B/thread)
    const int eA = tid * 16;            // halves
    const int rA = eA >> 5, cA = eA & 31;
    const int eB = tid * 8;             // halves
    const int rB = eB >> 5, cB = eB & 31;
    const size_t gA = (size_t)(rowM0 + rA) * D_DIM + cA;   // + k0 at issue time
    const size_t gB = (size_t)(colN0 + rB) * D_DIM + cB;

    // ---- prologue: stage K-step 0 into buffer 0 ----
    {
        async_copy_b128(&lA0[0][eA],     f0n + gA);
        async_copy_b128(&lA0[0][eA + 8], f0n + gA + 8);
        async_copy_b128(&lA1[0][eA],     f1n + gA);
        async_copy_b128(&lA1[0][eA + 8], f1n + gA + 8);
        async_copy_b128(&lB[0][eB],      tn  + gB);
    }

#pragma unroll 2
    for (int it = 0; it < KSTEPS; ++it) {
        const int buf = it & 1;
        const int k0  = it * BK;

        // Current buffer: my async writes done, then all waves' writes visible.
        wait_asynccnt0();
        __syncthreads();

        // Prefetch next K-step into the other buffer (its previous readers
        // all finished before the barrier above).
        if (it + 1 < KSTEPS) {
            const int nb = buf ^ 1;
            const int kn = k0 + BK;
            async_copy_b128(&lA0[nb][eA],     f0n + gA + kn);
            async_copy_b128(&lA0[nb][eA + 8], f0n + gA + kn + 8);
            async_copy_b128(&lA1[nb][eA],     f1n + gA + kn);
            async_copy_b128(&lA1[nb][eA + 8], f1n + gA + kn + 8);
            async_copy_b128(&lB[nb][eB],      tn  + gB + kn);
        }

        // ---- A fragments for this wave's M tile ----
        v8h a0lo = *(const v8h*)&lA0[buf][mrow * BK + kA_lo];
        v8h a0hi = *(const v8h*)&lA0[buf][mrow * BK + kA_lo + 16];
        v16h a0  = __builtin_shufflevector(a0lo, a0hi,
                       0,1,2,3,4,5,6,7,8,9,10,11,12,13,14,15);
        v8h a1lo = *(const v8h*)&lA1[buf][mrow * BK + kA_lo];
        v8h a1hi = *(const v8h*)&lA1[buf][mrow * BK + kA_lo + 16];
        v16h a1  = __builtin_shufflevector(a1lo, a1hi,
                       0,1,2,3,4,5,6,7,8,9,10,11,12,13,14,15);

#pragma unroll
        for (int nt = 0; nt < 4; ++nt) {
            const int brow = nt * 16 + (lane & 15);
            v8h blo = *(const v8h*)&lB[buf][brow * BK + kA_lo];
            v8h bhi = *(const v8h*)&lB[buf][brow * BK + kA_lo + 16];
            v16h bf = __builtin_shufflevector(blo, bhi,
                           0,1,2,3,4,5,6,7,8,9,10,11,12,13,14,15);
            acc0[nt] = __builtin_amdgcn_wmma_f32_16x16x32_f16(
                false, a0, false, bf, (short)0, acc0[nt], false, false);
            acc1[nt] = __builtin_amdgcn_wmma_f32_16x16x32_f16(
                false, a1, false, bf, (short)0, acc1[nt], false, false);
        }
    }

    // ---- fused margin loss on register-resident sim tiles ----
    // C/D layout: VGPR r, lane l: M = r + 8*(l>=16), N = l%16
    float part = 0.0f;
    const int mbase = rowM0 + wave * 16 + (lane >> 4) * 8;
    const int nlane = lane & 15;
#pragma unroll
    for (int nt = 0; nt < 4; ++nt) {
        const int n = colN0 + nt * 16 + nlane;
#pragma unroll
        for (int r = 0; r < 8; ++r) {
            const int m = mbase + r;
            const float lab = labels[(size_t)m * B_DIM + n];
            const float s0 = acc0[nt][r];
            const float s1 = acc1[nt][r];
            part += lab * (1.0f - s0) + (1.0f - lab) * fmaxf(s0 - MARGIN, 0.0f);
            part += lab * (1.0f - s1) + (1.0f - lab) * fmaxf(s1 - MARGIN, 0.0f);
        }
    }

    red[tid] = part;
    __syncthreads();
    for (int s = 128; s > 0; s >>= 1) {
        if (tid < s) red[tid] += red[tid + s];
        __syncthreads();
    }
    if (tid == 0)
        partials[blockIdx.y * gridDim.x + blockIdx.x] = red[0];
}

// ---------------------------------------------------------------------------
// Kernel 3: deterministic final reduction of block partials, scale by 1/B^2
// ---------------------------------------------------------------------------
__global__ void reduce_partials(const float* __restrict__ partials, int n,
                                float* __restrict__ out) {
    __shared__ float red[256];
    float s = 0.0f;
    for (int i = threadIdx.x; i < n; i += 256) s += partials[i];
    red[threadIdx.x] = s;
    __syncthreads();
    for (int k = 128; k > 0; k >>= 1) {
        if (threadIdx.x < k) red[threadIdx.x] += red[threadIdx.x + k];
        __syncthreads();
    }
    if (threadIdx.x == 0)
        out[0] = red[0] * (1.0f / ((float)B_DIM * (float)B_DIM));
}

// ---------------------------------------------------------------------------
extern "C" void kernel_launch(void* const* d_in, const int* in_sizes, int n_in,
                              void* d_out, int out_size, void* d_ws, size_t ws_size,
                              hipStream_t stream) {
    const float* f0     = (const float*)d_in[0];
    const float* f1     = (const float*)d_in[1];
    const float* t      = (const float*)d_in[2];
    const float* labels = (const float*)d_in[3];

    const size_t feat_elems = (size_t)B_DIM * D_DIM;
    _Float16* f0n = (_Float16*)d_ws;
    _Float16* f1n = f0n + feat_elems;
    _Float16* tn  = f1n + feat_elems;
    float* partials = (float*)(tn + feat_elems);

    // 1) row-normalize + f32->f16
    normalize_rows<<<3 * B_DIM, 256, 0, stream>>>(f0, f1, t, f0n, f1n, tn);

    // 2) fused dual WMMA GEMM + margin loss -> 2048 block partials
    dim3 grid(B_DIM / BN, B_DIM / BM);   // (64, 32)
    gemm_loss<<<grid, 256, 0, stream>>>(f0n, f1n, tn, labels, partials);

    // 3) final reduction -> scalar
    reduce_partials<<<1, 256, 0, stream>>>(partials, (B_DIM / BN) * (B_DIM / BM),
                                           (float*)d_out);
}
